// FlowSimCorrespondenceGenerationArch_21577915695510
// MI455X (gfx1250) — matrix operand order repaired and checked by reading, not compiled, not run on hardware
//
#include <hip/hip_runtime.h>
#include <hip/hip_bf16.h>
#include <math.h>

typedef __bf16 bf16_t;
typedef bf16_t v16bf __attribute__((ext_vector_type(16)));
typedef float  v8f   __attribute__((ext_vector_type(8)));

#define BATCH 8
#define CH    256
#define HH    64
#define WW    64
#define PH    62                    // H - 3 + 1
#define NPOS  (HH*WW)               // 4096
#define NPAT  (PH*PH)               // 3844

// ---- workspace layout (bytes) ----
static constexpr size_t WS_FEAT_IN  = 0;
static constexpr size_t WS_FEAT_REF = WS_FEAT_IN  + (size_t)BATCH*NPOS*CH*2;  // 16 MB each
static constexpr size_t WS_QIN      = WS_FEAT_REF + (size_t)BATCH*NPOS*CH*2;
static constexpr size_t WS_QREF     = WS_QIN      + (size_t)BATCH*NPOS*4;
static constexpr size_t WS_NORMIN   = WS_QREF     + (size_t)BATCH*NPOS*4;
static constexpr size_t WS_INVREF   = WS_NORMIN   + (size_t)BATCH*NPAT*4;
static constexpr size_t WS_MAXVAL   = WS_INVREF   + (size_t)BATCH*NPAT*4;
static constexpr size_t WS_MAXIDX   = WS_MAXVAL   + (size_t)BATCH*NPAT*4;

// ------------------------------------------------------------------
// 1) channel L2-normalize, transpose CHW(f32) -> HWC(bf16), and emit
//    per-position squared norm of the normalized vector (~1.0).
//    Block = 256 threads handling 32 spatial positions.
// ------------------------------------------------------------------
__global__ __launch_bounds__(256)
void normalize_kernel(const float* __restrict__ fin,
                      const float* __restrict__ fref,
                      bf16_t* __restrict__ featIn,
                      bf16_t* __restrict__ featRef,
                      float* __restrict__ qIn,
                      float* __restrict__ qRef)
{
    __shared__ float part[8][32];
    __shared__ float sinv[32];
    const int which = blockIdx.y;
    const float* src = which ? fref : fin;
    bf16_t* dst   = which ? featRef : featIn;
    float*  qdst  = which ? qRef    : qIn;

    const int posBase = blockIdx.x * 32;     // grid.x = BATCH*NPOS/32
    const int b  = posBase / NPOS;
    const int p  = posBase % NPOS;
    const int t  = threadIdx.x;
    const int pi = t & 31;
    const int ci = t >> 5;                   // 0..7

    const float* sp = src + (size_t)b * CH * NPOS + p + pi;  // [ch][pos]
    float ss = 0.f;
    for (int c = ci; c < CH; c += 8) {
        float v = sp[(size_t)c * NPOS];
        ss += v * v;
    }
    part[ci][pi] = ss;
    __syncthreads();
    if (t < 32) {
        float s = 0.f;
        #pragma unroll
        for (int k = 0; k < 8; ++k) s += part[k][t];
        float inv = 1.0f / fmaxf(sqrtf(s), 1e-12f);
        sinv[t] = inv;
        qdst[b * NPOS + p + t] = s * inv * inv;
    }
    __syncthreads();
    const float inv = sinv[pi];
    bf16_t* dp = dst + (size_t)(b * NPOS + p + pi) * CH;
    #pragma unroll
    for (int k = 0; k < 32; ++k) {
        int c = ci * 32 + k;
        dp[c] = (bf16_t)(sp[(size_t)c * NPOS] * inv);
    }
}

// ------------------------------------------------------------------
// 2) 3x3 patch norms from per-position squared norms.
// ------------------------------------------------------------------
__global__ __launch_bounds__(256)
void patchnorm_kernel(const float* __restrict__ qIn,
                      const float* __restrict__ qRef,
                      float* __restrict__ normIn,
                      float* __restrict__ invRef)
{
    int idx = blockIdx.x * blockDim.x + threadIdx.x;
    if (idx >= BATCH * NPAT) return;
    int b = idx / NPAT, n = idx % NPAT;
    int y = n / PH, x = n % PH;
    const float* qi = qIn  + b * NPOS;
    const float* qr = qRef + b * NPOS;
    float si = 0.f, sr = 0.f;
    #pragma unroll
    for (int i = 0; i < 3; ++i)
        #pragma unroll
        for (int j = 0; j < 3; ++j) {
            int sp = (y + i) * WW + (x + j);
            si += qi[sp]; sr += qr[sp];
        }
    normIn[idx] = sqrtf(si);
    invRef[idx] = 1.0f / (sqrtf(sr) + 1e-5f);
}

// ------------------------------------------------------------------
// 3) WMMA GEMM (M=N=3844, K=2304, bf16 x bf16 -> f32) with fused
//    per-row max/argmax over N.
//    Block = 128 threads (4 waves); each wave owns 2 M-tiles (32 rows)
//    and sweeps N four 16-col tiles at a time:
//      per k-step: 2 A frags (4 loads) + 4 B frags (8 loads) -> 8 WMMA
//      = 1.5 b128 loads per WMMA, 8 independent accumulators.
//    Fragments load straight from HWC bf16 feature maps (L2-resident).
// ------------------------------------------------------------------
__global__ __launch_bounds__(128)
void corr_argmax_kernel(const bf16_t* __restrict__ featIn,
                        const bf16_t* __restrict__ featRef,
                        const float* __restrict__ normIn,
                        const float* __restrict__ invRef,
                        float* __restrict__ maxVal,
                        int*   __restrict__ maxIdx)
{
    const int b    = blockIdx.y;
    const int wave = threadIdx.x >> 5;
    const int lane = threadIdx.x & 31;
    const int r    = lane & 15;      // A row / B col within tile
    const int s    = lane >> 4;      // half selector

    const int mt0 = blockIdx.x * 128 + wave * 32;
    const int mt1 = mt0 + 16;

    const int m0  = min(mt0 + r, NPAT - 1);
    const int m1  = min(mt1 + r, NPAT - 1);
    const int ay0 = m0 / PH, ax0 = m0 % PH;
    const int ay1 = m1 / PH, ax1 = m1 % PH;

    const uint4* fin  = (const uint4*)(featIn  + (size_t)b * NPOS * CH);
    const uint4* fref = (const uint4*)(featRef + (size_t)b * NPOS * CH);
    const float* ir   = invRef + b * NPAT;

    float bv[2][8];
    int   bi[2][8];
    #pragma unroll
    for (int mt = 0; mt < 2; ++mt)
        #pragma unroll
        for (int v = 0; v < 8; ++v) { bv[mt][v] = -INFINITY; bi[mt][v] = 0; }

    for (int nt = 0; nt < 241; nt += 4) {
        int nn[4], by[4], bx[4];
        #pragma unroll
        for (int q = 0; q < 4; ++q) {
            nn[q] = min((nt + q) * 16 + r, NPAT - 1);
            by[q] = nn[q] / PH;
            bx[q] = nn[q] % PH;
        }

        v8f acc[2][4];
        #pragma unroll
        for (int mt = 0; mt < 2; ++mt)
            #pragma unroll
            for (int q = 0; q < 4; ++q) acc[mt][q] = (v8f){};

        #pragma unroll 1
        for (int p = 0; p < 9; ++p) {
            const int iy = p / 3, jx = p % 3;
            // uint4 (16B = 8 bf16) index base of this spatial position's channel run
            const int aB0 = ((ay0 + iy) * WW + (ax0 + jx)) * (CH / 8);
            const int aB1 = ((ay1 + iy) * WW + (ax1 + jx)) * (CH / 8);
            int bB[4];
            #pragma unroll
            for (int q = 0; q < 4; ++q)
                bB[q] = ((by[q] + iy) * WW + (bx[q] + jx)) * (CH / 8);

            #pragma unroll
            for (int cb = 0; cb < 8; ++cb) {      // 8 x K32 blocks per patch pos
                union { v16bf v; uint4 u[2]; } a0, a1, bf[4];
                // A 16x32 bf16 layout: lane(r,s): K = s*8+{0..7} and s*8+16+{0..7}
                a0.u[0] = fin[aB0 + cb * 4 + s];
                a0.u[1] = fin[aB0 + cb * 4 + s + 2];
                a1.u[0] = fin[aB1 + cb * 4 + s];
                a1.u[1] = fin[aB1 + cb * 4 + s + 2];
                // B 32x16 bf16 layout: lane(c,s): K = s*8+{0..7} and s*8+16+{0..7}
                #pragma unroll
                for (int q = 0; q < 4; ++q) {
                    bf[q].u[0] = fref[bB[q] + cb * 4 + s];
                    bf[q].u[1] = fref[bB[q] + cb * 4 + s + 2];
                }
                #pragma unroll
                for (int q = 0; q < 4; ++q) {
                    acc[0][q] = __builtin_amdgcn_wmma_f32_16x16x32_bf16(
                        false, a0.v, false, bf[q].v, (short)0, acc[0][q], false, false);
                    acc[1][q] = __builtin_amdgcn_wmma_f32_16x16x32_bf16(
                        false, a1.v, false, bf[q].v, (short)0, acc[1][q], false, false);
                }
            }
        }

        // scale by ref patch norm, track running max/argmax
        // (q ascending keeps per-lane column order increasing -> first-hit ties)
        #pragma unroll
        for (int q = 0; q < 4; ++q) {
            const float sc = ir[nn[q]];
            #pragma unroll
            for (int v = 0; v < 8; ++v) {
                float t0 = acc[0][q][v] * sc;
                if (t0 > bv[0][v]) { bv[0][v] = t0; bi[0][v] = nn[q]; }
                float t1 = acc[1][q][v] * sc;
                if (t1 > bv[1][v]) { bv[1][v] = t1; bi[1][v] = nn[q]; }
            }
        }
    }

    // cross-lane reduction over the 16 columns held per row group
    // (xor masks 1,2,4,8 stay within same s-half; tie-break = lower index)
    #pragma unroll
    for (int mt = 0; mt < 2; ++mt) {
        #pragma unroll
        for (int v = 0; v < 8; ++v) {
            float val = bv[mt][v];
            int   idx = bi[mt][v];
            #pragma unroll
            for (int m = 8; m >= 1; m >>= 1) {
                float ov = __shfl_xor(val, m, 32);
                int   oi = __shfl_xor(idx, m, 32);
                if (ov > val || (ov == val && oi < idx)) { val = ov; idx = oi; }
            }
            if (r == 0) {
                int mrow = (mt ? mt1 : mt0) + s * 8 + v;   // C layout: M = v + s*8
                if (mrow < NPAT) {
                    maxVal[b * NPAT + mrow] = val / normIn[b * NPAT + mrow];
                    maxIdx[b * NPAT + mrow] = idx;
                }
            }
        }
    }
}

// ------------------------------------------------------------------
// 4) epilogue: pre_flow (B,64,64,2), pre_offset (B,9,64,64,2),
//    pre_similarity (B,1,64,64), concatenated flat in d_out.
// ------------------------------------------------------------------
__global__ __launch_bounds__(256)
void epilogue_kernel(const float* __restrict__ maxVal,
                     const int*   __restrict__ maxIdx,
                     float* __restrict__ out)
{
    int idx = blockIdx.x * blockDim.x + threadIdx.x;
    if (idx >= BATCH * NPOS) return;
    int b = idx / NPOS;
    int pos = idx % NPOS;
    int h = pos / WW, w = pos % WW;

    const float* mv = maxVal + b * NPAT;
    const int*   mi = maxIdx + b * NPAT;

    float* preFlow = out;                                   // (B,64,64,2)
    float* preOff  = out + (size_t)BATCH * NPOS * 2;        // (B,9,64,64,2)
    float* preSim  = preOff + (size_t)BATCH * 9 * NPOS * 2; // (B,1,64,64)

    // own flow (padded by 2 at bottom/right with zeros)
    float fx = 0.f, fy = 0.f;
    if (h < PH && w < PH) {
        int id = mi[h * PH + w];
        fx = (float)(id % PH) - (float)w;
        fy = (float)(id / PH) - (float)h;
    }
    size_t fb = ((size_t)(b * HH + h) * WW + w) * 2;
    preFlow[fb + 0] = fx;
    preFlow[fb + 1] = fy;

    // similarity: pad((1,1),(1,1))
    float sv = 0.f;
    if (h >= 1 && h <= PH && w >= 1 && w <= PH)
        sv = mv[(h - 1) * PH + (w - 1)];
    preSim[(size_t)(b * HH + h) * WW + w] = sv;

    // nine down-right shifts of padded flow (zero-fill top-left)
    #pragma unroll
    for (int t = 0; t < 9; ++t) {
        int i = t / 3, j = t % 3;
        int sy = h - i, sx = w - j;
        float gx = 0.f, gy = 0.f;
        if (sy >= 0 && sx >= 0 && sy < PH && sx < PH) {
            int id = mi[sy * PH + sx];
            gx = (float)(id % PH) - (float)sx;
            gy = (float)(id / PH) - (float)sy;
        }
        float* o = preOff + (((size_t)(b * 9 + t) * HH + h) * WW + w) * 2;
        o[0] = gx; o[1] = gy;
    }
}

extern "C" void kernel_launch(void* const* d_in, const int* in_sizes, int n_in,
                              void* d_out, int out_size, void* d_ws, size_t ws_size,
                              hipStream_t stream)
{
    const float* f1 = (const float*)d_in[0];
    const float* f2 = (const float*)d_in[1];
    char* ws = (char*)d_ws;

    bf16_t* featIn  = (bf16_t*)(ws + WS_FEAT_IN);
    bf16_t* featRef = (bf16_t*)(ws + WS_FEAT_REF);
    float*  qIn     = (float*)(ws + WS_QIN);
    float*  qRef    = (float*)(ws + WS_QREF);
    float*  normIn  = (float*)(ws + WS_NORMIN);
    float*  invRefA = (float*)(ws + WS_INVREF);
    float*  maxVal  = (float*)(ws + WS_MAXVAL);
    int*    maxIdx  = (int*)(ws + WS_MAXIDX);

    dim3 gN(BATCH * NPOS / 32, 2);
    normalize_kernel<<<gN, 256, 0, stream>>>(f1, f2, featIn, featRef, qIn, qRef);

    patchnorm_kernel<<<(BATCH * NPAT + 255) / 256, 256, 0, stream>>>(qIn, qRef, normIn, invRefA);

    dim3 gG((NPAT + 127) / 128, BATCH);   // 31 x 8 workgroups
    corr_argmax_kernel<<<gG, 128, 0, stream>>>(featIn, featRef, normIn, invRefA, maxVal, maxIdx);

    epilogue_kernel<<<(BATCH * NPOS + 255) / 256, 256, 0, stream>>>(maxVal, maxIdx, (float*)d_out);
}
